// MinimalResonanceLayer_16020228014207
// MI455X (gfx1250) — matrix operand order repaired and checked by reading, not compiled, not run on hardware
//
#include <hip/hip_runtime.h>
#include <cstdint>
#include <cstddef>

// ---------------------------------------------------------------------------
// Types for CDNA5 WMMA (wave32)
// ---------------------------------------------------------------------------
typedef __bf16 bf16;
typedef __attribute__((ext_vector_type(8)))  bf16  v8bf;
typedef __attribute__((ext_vector_type(16))) bf16  v16bf;
typedef __attribute__((ext_vector_type(8)))  float v8f;

// Problem constants (match reference)
constexpr int B  = 2, S = 2048, D = 1024, H = 16, HD = 64, DFF = 2048;
constexpr int BH = B * H;
constexpr size_t NE = (size_t)BH * S * HD;   // == B*S*D == 4,194,304
constexpr float MU = 1.0f, KC = 3.0f, DT = 0.02f, MIX = 0.3f;
constexpr float CA = 0.99500416527802576610f;   // cos(0.1)
constexpr float SA = 0.09983341664682815231f;   // sin(0.1)

#define DEVINL static __device__ __forceinline__

DEVINL v8f wmma_bf16(v16bf a, v16bf b, v8f c) {
  // v_wmma_f32_16x16x32_bf16  D = A(16x32) * B(32x16) + C(16x16 f32)
  return __builtin_amdgcn_wmma_f32_16x16x32_bf16(
      /*neg_a=*/false, a, /*neg_b=*/false, b,
      /*c_mod=*/(short)0, c, /*reuse_a=*/false, /*reuse_b=*/false);
}

// A-operand loader: lane holds M = lane&15; K base = kb (caller adds (lane>>4)*16).
// elements 0..15 -> K = kb+j (contiguous, matches B-operand loads)
DEVINL v16bf load_aop(const bf16* p) {
  v8bf lo = *reinterpret_cast<const v8bf*>(p);
  v8bf hi = *reinterpret_cast<const v8bf*>(p + 8);
  return __builtin_shufflevector(lo, hi, 0,1,2,3,4,5,6,7,8,9,10,11,12,13,14,15);
}

DEVINL float wredmax(float v) {
  #pragma unroll
  for (int o = 16; o > 0; o >>= 1) v = fmaxf(v, __shfl_xor(v, o, 32));
  return v;
}
DEVINL float wredsum(float v) {
  #pragma unroll
  for (int o = 16; o > 0; o >>= 1) v += __shfl_xor(v, o, 32);
  return v;
}

// ---------------------------------------------------------------------------
// K1/K8: LayerNorm (row of D=1024) -> bf16
// ---------------------------------------------------------------------------
__global__ void ln_bf16_kernel(const float* __restrict__ x, const float* __restrict__ g,
                               const float* __restrict__ be, bf16* __restrict__ out) {
  __shared__ float s1[256], s2[256];
  const int row = blockIdx.x, t = threadIdx.x;
  const float* xr = x + (size_t)row * D;
  float ls = 0.f, lq = 0.f;
  for (int i = t; i < D; i += 256) { float v = xr[i]; ls += v; lq += v * v; }
  s1[t] = ls; s2[t] = lq; __syncthreads();
  for (int o = 128; o > 0; o >>= 1) {
    if (t < o) { s1[t] += s1[t + o]; s2[t] += s2[t + o]; }
    __syncthreads();
  }
  const float mean = s1[0] * (1.0f / D);
  const float var  = s2[0] * (1.0f / D) - mean * mean;
  const float inv  = rsqrtf(var + 1e-5f);
  for (int i = t; i < D; i += 256)
    out[(size_t)row * D + i] = (bf16)((xr[i] - mean) * inv * g[i] + be[i]);
}

// ---------------------------------------------------------------------------
// K2: per-head Q,K,V projection (S x 64) = xh(S x 64) @ W(64 x 64) via WMMA.
//     Also initializes ODE state: a = v, b = 0.
// grid = (S/128, BH), block = 256 (8 waves, each a 16-row strip)
// ---------------------------------------------------------------------------
__global__ void qkv_kernel(const bf16* __restrict__ xn,
                           const float* __restrict__ Wq, const float* __restrict__ Wk,
                           const float* __restrict__ Wv,
                           bf16* __restrict__ qo, bf16* __restrict__ ko,
                           float* __restrict__ vo, float* __restrict__ aS,
                           float* __restrict__ bS) {
  __shared__ bf16 WT[3][64][64];            // [o][e(N)][d(K)]  (transposed for B-op reads)
  const int bh = blockIdx.y, b = bh >> 4, h = bh & 15;
  const int m0 = blockIdx.x * 128;
  const int t = threadIdx.x;
  const float* Ws[3] = { Wq + h * 4096, Wk + h * 4096, Wv + h * 4096 };
  #pragma unroll
  for (int j = 0; j < 48; j++) {            // 3*64*64 = 12288 = 48*256
    int e = t + j * 256, o = e >> 12, r = e & 4095, d = r >> 6, n = r & 63;
    WT[o][n][d] = (bf16)Ws[o][d * 64 + n];
  }
  __syncthreads();
  const int w = t >> 5, lane = t & 31, hi = lane >> 4, lm = lane & 15;
  const int mrow = m0 + w * 16;
  const bf16* xrow = xn + ((size_t)b * S + (mrow + lm)) * D + h * HD;
  for (int o = 0; o < 3; o++) {
    v8f acc[4] = {};
    #pragma unroll
    for (int kt = 0; kt < 2; kt++) {
      v16bf aop = load_aop(xrow + kt * 32 + hi * 16);
      #pragma unroll
      for (int nt = 0; nt < 4; nt++) {
        v16bf bop = *reinterpret_cast<const v16bf*>(&WT[o][nt * 16 + lm][kt * 32 + hi * 16]);
        acc[nt] = wmma_bf16(aop, bop, acc[nt]);
      }
    }
    #pragma unroll
    for (int nt = 0; nt < 4; nt++)
      #pragma unroll
      for (int r = 0; r < 8; r++) {
        const int mm = mrow + r + hi * 8, nn = nt * 16 + lm;
        const size_t idx = ((size_t)bh * S + mm) * HD + nn;
        const float val = acc[nt][r];
        if (o == 0)      qo[idx] = (bf16)val;
        else if (o == 1) ko[idx] = (bf16)val;
        else { vo[idx] = val; aS[idx] = val; bS[idx] = 0.0f; }
      }
  }
}

// ---------------------------------------------------------------------------
// K3: raw attention scores  A[s,t] = (q[s] . k[t]) / 8  -> bf16 (pre-softmax)
// grid = (S/128, S/128, BH), block = 256; wave strip = 16 rows x 128 cols
// ---------------------------------------------------------------------------
__global__ void scores_kernel(const bf16* __restrict__ q, const bf16* __restrict__ k,
                              bf16* __restrict__ A) {
  const int bh = blockIdx.z;
  const int m0 = blockIdx.x * 128, n0 = blockIdx.y * 128;
  const int t = threadIdx.x, w = t >> 5, lane = t & 31, hi = lane >> 4, lm = lane & 15;
  const int mrow = m0 + w * 16;
  const bf16* qrow  = q + ((size_t)bh * S + mrow + lm) * HD;
  const bf16* kbase = k + (size_t)bh * S * HD;
  v8f acc[8] = {};
  #pragma unroll
  for (int kt = 0; kt < 2; kt++) {
    v16bf aop = load_aop(qrow + kt * 32 + hi * 16);
    #pragma unroll
    for (int nt = 0; nt < 8; nt++) {
      const int tcol = n0 + nt * 16 + lm;                 // B-op: N = tcol, K = d
      v16bf bop = *reinterpret_cast<const v16bf*>(kbase + (size_t)tcol * HD + kt * 32 + hi * 16);
      acc[nt] = wmma_bf16(aop, bop, acc[nt]);
    }
  }
  bf16* Abh = A + (size_t)bh * S * S;
  #pragma unroll
  for (int nt = 0; nt < 8; nt++)
    #pragma unroll
    for (int r = 0; r < 8; r++)
      Abh[(size_t)(mrow + r + hi * 8) * S + (n0 + nt * 16 + lm)] = (bf16)(acc[nt][r] * 0.125f);
}

// ---------------------------------------------------------------------------
// K4: row softmax in place on bf16 A. One wave per row (S=2048 -> 64 elems/lane)
// ---------------------------------------------------------------------------
__global__ void softmax_kernel(bf16* __restrict__ A) {
  const int t = threadIdx.x, w = t >> 5, lane = t & 31;
  const size_t row = (size_t)blockIdx.x * 8 + w;
  bf16* Ar = A + row * S;
  float xv[64];
  float m = -1e30f;
  #pragma unroll
  for (int i = 0; i < 64; i++) { xv[i] = (float)Ar[i * 32 + lane]; m = fmaxf(m, xv[i]); }
  m = wredmax(m);
  float s = 0.f;
  #pragma unroll
  for (int i = 0; i < 64; i++) { float e = __expf(xv[i] - m); xv[i] = e; s += e; }
  const float inv = 1.0f / wredsum(s);
  #pragma unroll
  for (int i = 0; i < 64; i++) Ar[i * 32 + lane] = (bf16)(xv[i] * inv);
}

// ---------------------------------------------------------------------------
// K5: Z(Sx64) = A(SxS, bf16) @ X(Sx64, f32 converted to bf16 in LDS)
// grid = (S/128, BH), block = 256 (8 waves x 16-row strips); K loop of 64x32
// ---------------------------------------------------------------------------
__global__ void applyA_kernel(const bf16* __restrict__ A, const float* __restrict__ X,
                              float* __restrict__ Z) {
  __shared__ bf16 XT[64][32];               // [col(N)][krow(K)] transposed bf16 stage
  const int bh = blockIdx.y, m0 = blockIdx.x * 128;
  const int t = threadIdx.x, w = t >> 5, lane = t & 31, hi = lane >> 4, lm = lane & 15;
  const float* Xbh = X + (size_t)bh * S * HD;
  const bf16*  Arow = A + (size_t)bh * S * S + (size_t)(m0 + w * 16 + lm) * S;
  v8f acc[4] = {};
  for (int kt = 0; kt < 64; kt++) {
    __syncthreads();
    {   // stage X[kt*32 .. +32)[0..64) -> bf16 transposed (2048 elems, 8/thread)
      const int e0 = t * 8, kr = e0 >> 6, c = e0 & 63;
      v8f xv = *reinterpret_cast<const v8f*>(Xbh + (size_t)(kt * 32 + kr) * HD + c);
      #pragma unroll
      for (int j = 0; j < 8; j++) XT[c + j][kr] = (bf16)xv[j];
    }
    __syncthreads();
    if (kt + 1 < 64) __builtin_prefetch(Arow + (kt + 1) * 32, 0, 1);  // global_prefetch
    v16bf aop = load_aop(Arow + kt * 32 + hi * 16);
    #pragma unroll
    for (int nt = 0; nt < 4; nt++) {
      v16bf bop = *reinterpret_cast<const v16bf*>(&XT[nt * 16 + lm][hi * 16]);
      acc[nt] = wmma_bf16(aop, bop, acc[nt]);
    }
  }
  float* Zbh = Z + (size_t)bh * S * HD;
  const int mrow = m0 + w * 16;
  #pragma unroll
  for (int nt = 0; nt < 4; nt++)
    #pragma unroll
    for (int r = 0; r < 8; r++)
      Zbh[(size_t)(mrow + r + hi * 8) * HD + nt * 16 + lm] = acc[nt][r];
}

// ---------------------------------------------------------------------------
// K6/K7: Heun ODE elementwise halves.  i -> (bh = i>>17, d = i&63, h = bh&15)
// ---------------------------------------------------------------------------
__global__ void ode_f1_kernel(const float* __restrict__ a, const float* __restrict__ b,
                              const float* __restrict__ za, const float* __restrict__ zb,
                              const float* __restrict__ omega,
                              float* __restrict__ da1, float* __restrict__ db1,
                              float* __restrict__ ap, float* __restrict__ bp) {
  const size_t i = (size_t)blockIdx.x * 256 + threadIdx.x;
  const int d = (int)(i & 63), h = (int)((i >> 17) & 15);
  const float om = omega[h * 64 + d];
  const float av = a[i], bv = b[i], zav = za[i], zbv = zb[i];
  const float r2 = av * av + bv * bv;
  const float ra = zav * CA + zbv * SA, rb = zbv * CA - zav * SA;
  const float da = (MU - r2) * av - om * bv + KC * (ra - av);
  const float db = (MU - r2) * bv + om * av + KC * (rb - bv);
  da1[i] = da; db1[i] = db;
  ap[i] = av + DT * da; bp[i] = bv + DT * db;
}

__global__ void ode_f2_kernel(float* __restrict__ a, float* __restrict__ b,
                              const float* __restrict__ ap, const float* __restrict__ bp,
                              const float* __restrict__ za, const float* __restrict__ zb,
                              const float* __restrict__ da1, const float* __restrict__ db1,
                              const float* __restrict__ omega) {
  const size_t i = (size_t)blockIdx.x * 256 + threadIdx.x;
  const int d = (int)(i & 63), h = (int)((i >> 17) & 15);
  const float om = omega[h * 64 + d];
  const float av = ap[i], bv = bp[i], zav = za[i], zbv = zb[i];
  const float r2 = av * av + bv * bv;
  const float ra = zav * CA + zbv * SA, rb = zbv * CA - zav * SA;
  const float da2 = (MU - r2) * av - om * bv + KC * (ra - av);
  const float db2 = (MU - r2) * bv + om * av + KC * (rb - bv);
  a[i] += 0.5f * DT * (da1[i] + da2);
  b[i] += 0.5f * DT * (db1[i] + db2);
}

// ---------------------------------------------------------------------------
// K9: mixed = MIX*attn_v + (1-MIX)*a ; head_out = mixed @ Wo[h] ; x1 = x + scatter
// grid = (S/128, BH)
// ---------------------------------------------------------------------------
__global__ void merge_wo_kernel(const float* __restrict__ attnv, const float* __restrict__ aS,
                                const float* __restrict__ Wo, const float* __restrict__ x,
                                float* __restrict__ x1) {
  __shared__ bf16 WT[64][64];               // [e(N)][d(K)]
  const int bh = blockIdx.y, b = bh >> 4, h = bh & 15;
  const int m0 = blockIdx.x * 128;
  const int t = threadIdx.x, w = t >> 5, lane = t & 31, hi = lane >> 4, lm = lane & 15;
  const float* Wh = Wo + h * 4096;
  #pragma unroll
  for (int j = 0; j < 16; j++) {            // 4096 = 16*256
    int e = t + j * 256, d = e >> 6, n = e & 63;
    WT[n][d] = (bf16)Wh[d * 64 + n];
  }
  __syncthreads();
  const size_t rb0 = ((size_t)bh * S + m0 + w * 16 + lm) * HD;
  v8f acc[4] = {};
  #pragma unroll
  for (int kt = 0; kt < 2; kt++) {
    const int kb = kt * 32 + hi * 16;
    v8f v0 = *reinterpret_cast<const v8f*>(attnv + rb0 + kb);
    v8f v1 = *reinterpret_cast<const v8f*>(attnv + rb0 + kb + 8);
    v8f a0 = *reinterpret_cast<const v8f*>(aS + rb0 + kb);
    v8f a1 = *reinterpret_cast<const v8f*>(aS + rb0 + kb + 8);
    v16bf aop;
    #pragma unroll
    for (int j = 0; j < 8; j++) {
      aop[j]     = (bf16)(MIX * v0[j] + (1.0f - MIX) * a0[j]);
      aop[j + 8] = (bf16)(MIX * v1[j] + (1.0f - MIX) * a1[j]);
    }
    #pragma unroll
    for (int nt = 0; nt < 4; nt++) {
      v16bf bop = *reinterpret_cast<const v16bf*>(&WT[nt * 16 + lm][kt * 32 + hi * 16]);
      acc[nt] = wmma_bf16(aop, bop, acc[nt]);
    }
  }
  const int mrow = m0 + w * 16;
  #pragma unroll
  for (int nt = 0; nt < 4; nt++)
    #pragma unroll
    for (int r = 0; r < 8; r++) {
      const int mm = mrow + r + hi * 8, nn = nt * 16 + lm;
      const size_t gi = ((size_t)b * S + mm) * D + h * HD + nn;
      x1[gi] = x[gi] + acc[nt][r];
    }
}

// ---------------------------------------------------------------------------
// K10: hg = gelu(x1n @ W1 + bf1)   M=B*S=4096, N=DFF=2048, K=D=1024
// grid = (M/128, N/128), block = 256; W1 tile staged transposed bf16 in LDS
// ---------------------------------------------------------------------------
__global__ void ffn1_kernel(const bf16* __restrict__ xn, const float* __restrict__ W1,
                            const float* __restrict__ bf1, bf16* __restrict__ hg) {
  __shared__ bf16 WT[128][32];              // [n][k]
  const int m0 = blockIdx.x * 128, n0 = blockIdx.y * 128;
  const int t = threadIdx.x, w = t >> 5, lane = t & 31, hi = lane >> 4, lm = lane & 15;
  const bf16* arow = xn + (size_t)(m0 + w * 16 + lm) * D;
  v8f acc[8] = {};
  for (int kt = 0; kt < 32; kt++) {
    __syncthreads();
    {   // stage W1[kt*32..+32)[n0..n0+128) -> transposed bf16 (16/thread)
      const int kr = t >> 3, c0 = (t & 7) * 16;
      const float* src = W1 + (size_t)(kt * 32 + kr) * DFF + n0 + c0;
      v8f w0 = *reinterpret_cast<const v8f*>(src);
      v8f w1 = *reinterpret_cast<const v8f*>(src + 8);
      #pragma unroll
      for (int j = 0; j < 8; j++) { WT[c0 + j][kr] = (bf16)w0[j]; WT[c0 + 8 + j][kr] = (bf16)w1[j]; }
    }
    __syncthreads();
    v16bf aop = load_aop(arow + kt * 32 + hi * 16);
    #pragma unroll
    for (int nt = 0; nt < 8; nt++) {
      v16bf bop = *reinterpret_cast<const v16bf*>(&WT[nt * 16 + lm][hi * 16]);
      acc[nt] = wmma_bf16(aop, bop, acc[nt]);
    }
  }
  const int mrow = m0 + w * 16;
  #pragma unroll
  for (int nt = 0; nt < 8; nt++)
    #pragma unroll
    for (int r = 0; r < 8; r++) {
      const int nn = n0 + nt * 16 + lm;
      float v = acc[nt][r] + bf1[nn];
      float g = 0.5f * v * (1.0f + tanhf(0.7978845608f * (v + 0.044715f * v * v * v)));
      hg[(size_t)(mrow + r + hi * 8) * DFF + nn] = (bf16)g;
    }
}

// ---------------------------------------------------------------------------
// K11: out = x1 + hg @ W2 + bf2   M=4096, N=D=1024, K=DFF=2048
// ---------------------------------------------------------------------------
__global__ void ffn2_kernel(const bf16* __restrict__ hg, const float* __restrict__ W2,
                            const float* __restrict__ bf2, const float* __restrict__ x1,
                            float* __restrict__ out) {
  __shared__ bf16 WT[128][32];
  const int m0 = blockIdx.x * 128, n0 = blockIdx.y * 128;
  const int t = threadIdx.x, w = t >> 5, lane = t & 31, hi = lane >> 4, lm = lane & 15;
  const bf16* arow = hg + (size_t)(m0 + w * 16 + lm) * DFF;
  v8f acc[8] = {};
  for (int kt = 0; kt < 64; kt++) {
    __syncthreads();
    {
      const int kr = t >> 3, c0 = (t & 7) * 16;
      const float* src = W2 + (size_t)(kt * 32 + kr) * D + n0 + c0;
      v8f w0 = *reinterpret_cast<const v8f*>(src);
      v8f w1 = *reinterpret_cast<const v8f*>(src + 8);
      #pragma unroll
      for (int j = 0; j < 8; j++) { WT[c0 + j][kr] = (bf16)w0[j]; WT[c0 + 8 + j][kr] = (bf16)w1[j]; }
    }
    __syncthreads();
    v16bf aop = load_aop(arow + kt * 32 + hi * 16);
    #pragma unroll
    for (int nt = 0; nt < 8; nt++) {
      v16bf bop = *reinterpret_cast<const v16bf*>(&WT[nt * 16 + lm][hi * 16]);
      acc[nt] = wmma_bf16(aop, bop, acc[nt]);
    }
  }
  const int mrow = m0 + w * 16;
  #pragma unroll
  for (int nt = 0; nt < 8; nt++)
    #pragma unroll
    for (int r = 0; r < 8; r++) {
      const int mm = mrow + r + hi * 8, nn = n0 + nt * 16 + lm;
      const size_t gi = (size_t)mm * D + nn;
      out[gi] = acc[nt][r] + bf2[nn] + x1[gi];
    }
}

// ---------------------------------------------------------------------------
// Host-side orchestration
// ---------------------------------------------------------------------------
extern "C" void kernel_launch(void* const* d_in, const int* in_sizes, int n_in,
                              void* d_out, int out_size, void* d_ws, size_t ws_size,
                              hipStream_t stream) {
  (void)in_sizes; (void)n_in; (void)out_size; (void)ws_size;
  const float* x     = (const float*)d_in[0];
  const float* Wq    = (const float*)d_in[1];
  const float* Wk    = (const float*)d_in[2];
  const float* Wv    = (const float*)d_in[3];
  const float* Wo    = (const float*)d_in[4];
  const float* omega = (const float*)d_in[5];
  const float* g1    = (const float*)d_in[6];
  const float* be1   = (const float*)d_in[7];
  const float* g2    = (const float*)d_in[8];
  const float* be2   = (const float*)d_in[9];
  const float* W1    = (const float*)d_in[10];
  const float* bf1   = (const float*)d_in[11];
  const float* W2    = (const float*)d_in[12];
  const float* bf2   = (const float*)d_in[13];
  float* out = (float*)d_out;

  char* ws = (char*)d_ws;
  size_t off = 0;
  auto alloc = [&](size_t bytes) { void* p = ws + off; off += (bytes + 255) & ~(size_t)255; return p; };

  bf16*  xn    = (bf16*)alloc(NE * 2);
  bf16*  qb    = (bf16*)alloc(NE * 2);
  bf16*  kb    = (bf16*)alloc(NE * 2);
  float* vb    = (float*)alloc(NE * 4);
  bf16*  A     = (bf16*)alloc((size_t)BH * S * S * 2);   // 256 MB
  float* aS    = (float*)alloc(NE * 4);
  float* bS    = (float*)alloc(NE * 4);
  float* da1   = (float*)alloc(NE * 4);
  float* db1   = (float*)alloc(NE * 4);
  float* apB   = (float*)alloc(NE * 4);
  float* bpB   = (float*)alloc(NE * 4);
  float* za    = (float*)alloc(NE * 4);
  float* zb    = (float*)alloc(NE * 4);
  float* attnv = (float*)alloc(NE * 4);
  float* x1    = (float*)alloc(NE * 4);
  bf16*  x1n   = (bf16*)alloc(NE * 2);
  bf16*  hg    = (bf16*)alloc((size_t)B * S * DFF * 2);

  const int EW = (int)(NE / 256);           // elementwise grid

  // 1) LN1 -> bf16
  ln_bf16_kernel<<<B * S, 256, 0, stream>>>(x, g1, be1, xn);
  // 2) QKV projections + ODE state init (a=v, b=0)
  qkv_kernel<<<dim3(S / 128, BH), 256, 0, stream>>>(xn, Wq, Wk, Wv, qb, kb, vb, aS, bS);
  // 3) raw scores, 4) softmax in place
  scores_kernel<<<dim3(S / 128, S / 128, BH), 256, 0, stream>>>(qb, kb, A);
  softmax_kernel<<<(BH * S) / 8, 256, 0, stream>>>(A);
  // 5) attn_v = A @ v (original v; independent of ODE)
  applyA_kernel<<<dim3(S / 128, BH), 256, 0, stream>>>(A, vb, attnv);
  // 6) Heun integration: 5 steps, 4 A-applications + 2 elementwise per step
  for (int step = 0; step < 5; step++) {
    applyA_kernel<<<dim3(S / 128, BH), 256, 0, stream>>>(A, aS, za);
    applyA_kernel<<<dim3(S / 128, BH), 256, 0, stream>>>(A, bS, zb);
    ode_f1_kernel<<<EW, 256, 0, stream>>>(aS, bS, za, zb, omega, da1, db1, apB, bpB);
    applyA_kernel<<<dim3(S / 128, BH), 256, 0, stream>>>(A, apB, za);
    applyA_kernel<<<dim3(S / 128, BH), 256, 0, stream>>>(A, bpB, zb);
    ode_f2_kernel<<<EW, 256, 0, stream>>>(aS, bS, apB, bpB, za, zb, da1, db1, omega);
  }
  // 7) mix + Wo + residual -> x1
  merge_wo_kernel<<<dim3(S / 128, BH), 256, 0, stream>>>(attnv, aS, Wo, x, x1);
  // 8) LN2 -> bf16
  ln_bf16_kernel<<<B * S, 256, 0, stream>>>(x1, g2, be2, x1n);
  // 9) FFN up + gelu
  ffn1_kernel<<<dim3((B * S) / 128, DFF / 128), 256, 0, stream>>>(x1n, W1, bf1, hg);
  // 10) FFN down + bias + residual -> out
  ffn2_kernel<<<dim3((B * S) / 128, D / 128), 256, 0, stream>>>(hg, W2, bf2, x1, out);
}
